// SimilarityTreeLSTM_13795434954929
// MI455X (gfx1250) — compile-verified, matrix operand order checked by reading, not run on hardware
//
#include <hip/hip_runtime.h>
#include <stdint.h>

// ---------------- types ----------------
typedef __attribute__((ext_vector_type(16))) __bf16        v16bf;
typedef __attribute__((ext_vector_type(8)))  float         v8f;
typedef __attribute__((ext_vector_type(4)))  unsigned int  v4u;
typedef __attribute__((ext_vector_type(8)))  int           v8i;
typedef __attribute__((ext_vector_type(4)))  int           v4i;

typedef __attribute__((address_space(3))) unsigned short   lds_u16;
typedef __attribute__((address_space(3))) const v4u        lds_cv4u;

union Frag32B { v4u q[2]; v16bf v; };   // 32 bytes = 16 bf16 = one wmma A/B fragment per lane

static __device__ __forceinline__ unsigned short f32_to_bf16(float f) {
    uint32_t u = __float_as_uint(f);
    uint32_t r = u + 0x7FFFu + ((u >> 16) & 1u);   // round to nearest even
    return (unsigned short)(r >> 16);
}
static __device__ __forceinline__ float sigf(float x) { return 1.f / (1.f + __expf(-x)); }

// ---------------------------------------------------------------------------
// Pack a row-major f32 weight W[K][Nc] into bf16 WMMA-B fragment order.
// One wave per 32x16 (KxN) tile, per-lane 16 bf16 stored contiguously (32B):
//   col = nt*16 + lane%16 ; k(lane,e) = kt*32 + (lane>=16?8:0) + (e<8?e:e+8)
// Tile order in memory: tile(kt,nt) at ((kt*ntiles)+nt)*512 elements.
// ---------------------------------------------------------------------------
__global__ __launch_bounds__(32)
void pack_b_kernel(const float* __restrict__ W, unsigned short* __restrict__ out, int Nc) {
    int nt = blockIdx.x, kt = blockIdx.y;
    int lane = threadIdx.x;
    int col = nt * 16 + (lane & 15);
    int khalf = (lane >= 16) ? 8 : 0;
    unsigned short* dst = out + ((size_t)kt * gridDim.x + nt) * 512 + (size_t)lane * 16;
#pragma unroll
    for (int e = 0; e < 16; ++e) {
        int k = kt * 32 + khalf + ((e < 8) ? e : e + 8);
        dst[e] = f32_to_bf16(W[(size_t)k * Nc + col]);
    }
}

// ---------------------------------------------------------------------------
// Embedding gather + f32 -> bf16
// ---------------------------------------------------------------------------
__global__ void gather_embed_kernel(const int* __restrict__ ops, const float* __restrict__ emb,
                                    unsigned short* __restrict__ xbf, int n, int dim) {
    int t = blockIdx.x * blockDim.x + threadIdx.x;
    if (t >= n * dim) return;
    int row = t / dim, d = t - row * dim;
    xbf[t] = f32_to_bf16(emb[(size_t)ops[row] * dim + d]);
}

// ---------------------------------------------------------------------------
// C[M][Nc] = A[M][512](bf16) x Bp(packed bf16) + bias (+ C if accumulate)
// Block = 4 waves, 128(M) x 64(N) macro-tile. The shared B panel
// (K=512 x 64 cols = 64KB) is DMA'd into LDS once per block by the Tensor
// Data Mover (2D tile: 16 rows of 4KB, row stride = ntiles*1KB), then the
// k-loop reads B fragments from LDS (ds_load_b128) and A from global with
// CLAMPED row indices (no EXEC churn in the hot loop).
// Each wave: 2 M-tiles x 4 N-tiles -> 8 wmma per k-tile.
//
// The TDM builtin carries no pointer operand tying it to smemB, and a bare
// "memory" clobber is not enough (smemB never escapes, so provenance-based
// AA still proves it unwritten and folds the loads to undef — seen in
// rounds 2-3). Fix: LAUNDER the addrspace(3) base pointer through an asm
// operand so the address escapes and the loads must really be emitted.
// ---------------------------------------------------------------------------
#define GEMM_KT 16   // K = 512 = 16 k-tiles of 32

__global__ __launch_bounds__(128)
void gemm_bf16_wmma_tdm_kernel(const unsigned short* __restrict__ A,
                               const unsigned short* __restrict__ Bp,
                               const float* __restrict__ bias,
                               float* __restrict__ C,
                               int M, int Nc, int accumulate) {
    __shared__ unsigned short smemB[GEMM_KT * 4 * 512];   // 64 KB, LDS offset 0

    const int lane = threadIdx.x & 31;
    const int warp = threadIdx.x >> 5;                     // 0..3
    const int col = lane & 15;
    const int hi  = lane >> 4;                             // 0 or 1
    const int ntilesAll = Nc >> 4;

    // ---- TDM: stream the B panel for this n-block into LDS (wave 0 only) ----
    if (warp == 0) {
        unsigned long long gaddr =
            (unsigned long long)(uintptr_t)Bp + (unsigned long long)blockIdx.x * 4u * 1024u;
        unsigned int dim0 = (unsigned)ntilesAll * 128u;    // row length, 8-byte units
        v4u g0;
        g0[0] = 1u;                                        // count=1 (valid descriptor)
        g0[1] = 0u;                                        // lds_addr = 0 (smemB at offset 0)
        g0[2] = (unsigned)gaddr;                           // global_addr[31:0]
        g0[3] = (unsigned)((gaddr >> 32) & 0x1FFFFFFu) | (2u << 30);  // addr[56:32] | type=2
        v8i g1;
        g1[0] = (int)(3u << 16);                           // data_size = 8B, no multicast
        g1[1] = (int)((dim0 & 0xFFFFu) << 16);             // tensor_dim0[15:0]
        g1[2] = (int)(((dim0 >> 16) & 0xFFFFu) | ((unsigned)GEMM_KT << 16)); // dim0 hi | tensor_dim1 lo
        g1[3] = (int)(512u << 16);                         // tensor_dim1 hi=0 | tile_dim0=512 (4KB/8B)
        g1[4] = (int)(GEMM_KT & 0xFFFFu);                  // tile_dim1=16, tile_dim2=0
        g1[5] = (int)dim0;                                 // tensor_dim0_stride[31:0]
        g1[6] = 0;                                         // stride0 hi | stride1 lo
        g1[7] = 0;
        v4i z4; z4[0] = 0; z4[1] = 0; z4[2] = 0; z4[3] = 0;
#if __clang_major__ >= 23
        v8i z8;
#pragma unroll
        for (int q = 0; q < 8; ++q) z8[q] = 0;
        __builtin_amdgcn_tensor_load_to_lds(g0, g1, z4, z4, z8, 0);
#else
        __builtin_amdgcn_tensor_load_to_lds(g0, g1, z4, z4, 0);
#endif
        __builtin_amdgcn_s_wait_tensorcnt(0);
    }
    __syncthreads();

    // Launder the LDS base pointer: escapes smemB's address and makes it an
    // asm-defined value, so the B-fragment loads below cannot be folded away
    // even though the TDM write is invisible to alias analysis.
    lds_u16* sb = (lds_u16*)&smemB[0];
    asm volatile("" : "+v"(sb));

    // ---- accumulator init: bias (+ existing C when accumulating) ----
    v8f acc[2][4];
#pragma unroll
    for (int i = 0; i < 2; ++i)
#pragma unroll
        for (int j = 0; j < 4; ++j) {
            int cb = blockIdx.x * 64 + j * 16 + col;
            float b = bias[cb];
#pragma unroll
            for (int e = 0; e < 8; ++e) {
                float v = b;
                if (accumulate) {
                    int r = blockIdx.y * 128 + (warp * 2 + i) * 16 + e + hi * 8;
                    int rc = r < M ? r : (M - 1);
                    float cv = C[(size_t)rc * Nc + cb];
                    if (r < M) v += cv;
                }
                acc[i][j][e] = v;
            }
        }

    // ---- hoisted A-row pointers (clamped rows: no EXEC churn in k-loop) ----
    const unsigned short* aptr[2];
#pragma unroll
    for (int i = 0; i < 2; ++i) {
        int r = blockIdx.y * 128 + (warp * 2 + i) * 16 + col;
        r = r < M ? r : (M - 1);
        aptr[i] = A + (size_t)r * 512 + hi * 8;
    }

    // ---- k-loop: A from global, B from LDS (ds_load_b128) ----
    for (int kt = 0; kt < GEMM_KT; ++kt) {
        Frag32B a[2], b[4];
#pragma unroll
        for (int i = 0; i < 2; ++i) {
            const unsigned short* p = aptr[i] + kt * 32;
            a[i].q[0] = *(const v4u*)(p);
            a[i].q[1] = *(const v4u*)(p + 16);
        }
#pragma unroll
        for (int j = 0; j < 4; ++j) {
            const lds_cv4u* sp = (const lds_cv4u*)(sb + kt * 2048 + j * 512 + lane * 16);
            b[j].q[0] = sp[0];
            b[j].q[1] = sp[1];
        }
#pragma unroll
        for (int i = 0; i < 2; ++i)
#pragma unroll
            for (int j = 0; j < 4; ++j)
                acc[i][j] = __builtin_amdgcn_wmma_f32_16x16x32_bf16(
                    false, a[i].v, false, b[j].v, (short)0, acc[i][j], false, false);
    }

    // ---- store (guard only here; EXEC was full through all WMMAs) ----
#pragma unroll
    for (int i = 0; i < 2; ++i)
#pragma unroll
        for (int j = 0; j < 4; ++j) {
            int cb = blockIdx.x * 64 + j * 16 + col;
#pragma unroll
            for (int e = 0; e < 8; ++e) {
                int r = blockIdx.y * 128 + (warp * 2 + i) * 16 + e + hi * 8;
                if (r < M) C[(size_t)r * Nc + cb] = acc[i][j][e];
            }
        }
}

// c[N] = h[N] = 1.0 (virtual "null child" row); other rows written before read.
__global__ void init_last_row_kernel(float* c, float* h, int N, int dim) {
    int d = blockIdx.x * blockDim.x + threadIdx.x;
    if (d < dim) { c[(size_t)N * dim + d] = 1.f; h[(size_t)N * dim + d] = 1.f; }
}

// Gather children h, apply mask, emit bf16 ch (rows*4 x dim) and bf16 h_sum (rows x dim).
__global__ void level_gather_kernel(const int* __restrict__ child_idx, const int* __restrict__ child_mask,
                                    const float* __restrict__ h,
                                    unsigned short* __restrict__ hsum_bf, unsigned short* __restrict__ ch_bf,
                                    int s, int rows, int dim) {
    int t = blockIdx.x * blockDim.x + threadIdx.x;
    if (t >= rows * dim) return;
    int r = t / dim, d = t - r * dim;
    int node = s + r;
    float sum = 0.f;
#pragma unroll
    for (int j = 0; j < 4; ++j) {
        int idx = child_idx[node * 4 + j];
        float m = (float)child_mask[node * 4 + j];
        float hv = h[(size_t)idx * dim + d] * m;
        ch_bf[(size_t)(r * 4 + j) * dim + d] = f32_to_bf16(hv);
        sum += hv;
    }
    hsum_bf[(size_t)r * dim + d] = f32_to_bf16(sum);
}

// Gates + state update.
__global__ void level_elem_kernel(const int* __restrict__ child_idx, const int* __restrict__ child_mask,
                                  const float* __restrict__ iou, const float* __restrict__ fx,
                                  const float* __restrict__ f_part,
                                  float* __restrict__ c, float* __restrict__ h,
                                  int s, int rows, int dim) {
    int t = blockIdx.x * blockDim.x + threadIdx.x;
    if (t >= rows * dim) return;
    int r = t / dim, d = t - r * dim;
    int node = s + r;
    const float* irow = iou + (size_t)node * 3 * dim;
    float iv = sigf(irow[d]);
    float ov = sigf(irow[dim + d]);
    float uv = tanhf(irow[2 * dim + d]);
    float fxv = fx[(size_t)node * dim + d];
    float fc = 0.f;
#pragma unroll
    for (int j = 0; j < 4; ++j) {
        int idx = child_idx[node * 4 + j];
        float m = (float)child_mask[node * 4 + j];
        float fv = sigf(f_part[(size_t)(r * 4 + j) * dim + d] + fxv);
        fc += fv * c[(size_t)idx * dim + d] * m;
    }
    float cn = iv * uv + fc;
    c[(size_t)node * dim + d] = cn;
    h[(size_t)node * dim + d] = ov * tanhf(cn);
}

__global__ void copy_root_kernel(const float* __restrict__ c, float* __restrict__ dst, int dim) {
    int d = blockIdx.x * blockDim.x + threadIdx.x;
    if (d < dim) dst[d] = c[d];
}

// Similarity head: tiny, single block, plain VALU.
__global__ __launch_bounds__(256)
void final_kernel(const float* __restrict__ lc, const float* __restrict__ rc,
                  const float* __restrict__ Wwh, const float* __restrict__ bwh,
                  const float* __restrict__ Wwp, const float* __restrict__ bwp,
                  float* __restrict__ out) {
    __shared__ float vec[1024];
    __shared__ float hv[512];
    __shared__ float logits[2];
    int tid = threadIdx.x;
    for (int k = tid; k < 512; k += 256) {
        float a = lc[k], b = rc[k];
        vec[k] = a * b;
        vec[512 + k] = fabsf(a - b);
    }
    __syncthreads();
    for (int j = tid; j < 512; j += 256) {
        float acc = bwh[j];
        for (int k = 0; k < 1024; ++k) acc += vec[k] * Wwh[(size_t)k * 512 + j];
        hv[j] = sigf(acc);
    }
    __syncthreads();
    if (tid < 2) {
        float acc = bwp[tid];
        for (int j = 0; j < 512; ++j) acc += hv[j] * Wwp[j * 2 + tid];
        logits[tid] = acc;
    }
    __syncthreads();
    if (tid == 0) {
        float m = fmaxf(logits[0], logits[1]);
        float e0 = __expf(logits[0] - m), e1 = __expf(logits[1] - m);
        float inv = 1.f / (e0 + e1);
        out[0] = e0 * inv;
        out[1] = e1 * inv;
    }
}

// ---------------------------------------------------------------------------
extern "C" void kernel_launch(void* const* d_in, const int* in_sizes, int n_in,
                              void* d_out, int out_size, void* d_ws, size_t ws_size,
                              hipStream_t stream) {
    const int*   lops       = (const int*)d_in[0];
    const int*   rops       = (const int*)d_in[1];
    const int*   child_idx  = (const int*)d_in[2];
    const int*   child_mask = (const int*)d_in[3];
    const float* emb        = (const float*)d_in[4];
    const float* Wx         = (const float*)d_in[5];
    const float* bx         = (const float*)d_in[6];
    const float* Wh         = (const float*)d_in[7];
    const float* bh         = (const float*)d_in[8];
    const float* Wfx        = (const float*)d_in[9];
    const float* bfx        = (const float*)d_in[10];
    const float* Wfh        = (const float*)d_in[11];
    const float* bfh        = (const float*)d_in[12];
    const float* Wwh        = (const float*)d_in[13];
    const float* bwh        = (const float*)d_in[14];
    const float* Wwp        = (const float*)d_in[15];
    const float* bwp        = (const float*)d_in[16];
    (void)in_sizes; (void)n_in; (void)out_size; (void)ws_size;

    const int N = 5461;
    static const int LS[8] = {0, 1, 5, 21, 85, 341, 1365, 5461};

    char* ws = (char*)d_ws;
    size_t off = 0;
    auto alloc = [&](size_t bytes) -> char* {
        char* p = ws + off;
        off = (off + bytes + 255) & ~(size_t)255;
        return p;
    };
    unsigned short* WxP    = (unsigned short*)alloc((size_t)512 * 1536 * 2);
    unsigned short* WhP    = (unsigned short*)alloc((size_t)512 * 1536 * 2);
    unsigned short* WfxP   = (unsigned short*)alloc((size_t)512 * 512 * 2);
    unsigned short* WfhP   = (unsigned short*)alloc((size_t)512 * 512 * 2);
    unsigned short* xbf    = (unsigned short*)alloc((size_t)N * 512 * 2);
    float*          ioux   = (float*)alloc((size_t)N * 1536 * 4);
    float*          fx     = (float*)alloc((size_t)N * 512 * 4);
    float*          cbuf   = (float*)alloc((size_t)(N + 1) * 512 * 4);
    float*          hbuf   = (float*)alloc((size_t)(N + 1) * 512 * 4);
    unsigned short* hsumbf = (unsigned short*)alloc((size_t)4096 * 512 * 2);
    unsigned short* chbf   = (unsigned short*)alloc((size_t)16384 * 512 * 2);
    float*          f_part = (float*)alloc((size_t)16384 * 512 * 4);
    float*          rootL  = (float*)alloc((size_t)512 * 4);

    // Pack weights into WMMA B-fragment order (bf16).
    pack_b_kernel<<<dim3(1536 / 16, 512 / 32), 32, 0, stream>>>(Wx, WxP, 1536);
    pack_b_kernel<<<dim3(1536 / 16, 512 / 32), 32, 0, stream>>>(Wh, WhP, 1536);
    pack_b_kernel<<<dim3(512 / 16, 512 / 32), 32, 0, stream>>>(Wfx, WfxP, 512);
    pack_b_kernel<<<dim3(512 / 16, 512 / 32), 32, 0, stream>>>(Wfh, WfhP, 512);

    for (int tree = 0; tree < 2; ++tree) {
        const int* ops = tree ? rops : lops;
        int total = N * 512;

        gather_embed_kernel<<<(total + 255) / 256, 256, 0, stream>>>(ops, emb, xbf, N, 512);

        // iou_x = x @ Wx + bx   (N x 1536)
        gemm_bf16_wmma_tdm_kernel<<<dim3(1536 / 64, (N + 127) / 128), 128, 0, stream>>>(
            xbf, WxP, bx, ioux, N, 1536, 0);
        // fx = x @ Wfx + bfx    (N x 512)
        gemm_bf16_wmma_tdm_kernel<<<dim3(512 / 64, (N + 127) / 128), 128, 0, stream>>>(
            xbf, WfxP, bfx, fx, N, 512, 0);

        init_last_row_kernel<<<2, 256, 0, stream>>>(cbuf, hbuf, N, 512);

        for (int lvl = 6; lvl >= 0; --lvl) {
            int s = LS[lvl], e = LS[lvl + 1], rows = e - s;
            int t1 = rows * 512;

            level_gather_kernel<<<(t1 + 255) / 256, 256, 0, stream>>>(
                child_idx, child_mask, hbuf, hsumbf, chbf, s, rows, 512);

            // iou[s:e] += h_sum @ Wh + bh   (accumulate in place, ldc = 1536)
            gemm_bf16_wmma_tdm_kernel<<<dim3(1536 / 64, (rows + 127) / 128), 128, 0, stream>>>(
                hsumbf, WhP, bh, ioux + (size_t)s * 1536, rows, 1536, 1);

            // f_part = ch @ Wfh + bfh       ((rows*4) x 512)
            gemm_bf16_wmma_tdm_kernel<<<dim3(512 / 64, (rows * 4 + 127) / 128), 128, 0, stream>>>(
                chbf, WfhP, bfh, f_part, rows * 4, 512, 0);

            level_elem_kernel<<<(t1 + 255) / 256, 256, 0, stream>>>(
                child_idx, child_mask, ioux, fx, f_part, cbuf, hbuf, s, rows, 512);
        }

        if (tree == 0)
            copy_root_kernel<<<2, 256, 0, stream>>>(cbuf, rootL, 512);
    }

    final_kernel<<<1, 256, 0, stream>>>(rootL, cbuf, Wwh, bwh, Wwp, bwp, (float*)d_out);
}
